// Critic_3032246911586
// MI455X (gfx1250) — compile-verified, matrix operand order checked by reading, not run on hardware
//
#include <hip/hip_runtime.h>
#include <cstdint>
#include <cstddef>


typedef __attribute__((ext_vector_type(16))) __bf16 v16bf;
typedef __attribute__((ext_vector_type(8)))  float  v8f;
typedef __attribute__((ext_vector_type(8)))  int    v8i_t;

#define DEVFN __device__ __forceinline__

DEVFN unsigned short f2bf(float x) {
    union { float f; unsigned u; } c; c.f = x;
    unsigned u = c.u;
    unsigned r = (u + 0x7FFFu + ((u >> 16) & 1u)) >> 16;
    return (unsigned short)r;
}
DEVFN float bf2f(unsigned short h) {
    union { unsigned u; float f; } c; c.u = ((unsigned)h) << 16;
    return c.f;
}
DEVFN float sigm(float x) { return 1.0f / (1.0f + __expf(-x)); }

DEVFN unsigned char f2fp8(float x) {
    union { float f; unsigned u; } c; c.f = x;
    unsigned u = c.u;
    unsigned s = (u >> 24) & 0x80u;
    unsigned fexp = (u >> 23) & 0xFFu;
    unsigned man = u & 0x7FFFFFu;
    if ((u & 0x7FFFFFFFu) == 0) return (unsigned char)s;
    if (fexp == 0xFFu) return (unsigned char)(s | 0x7E);
    int e8 = (int)fexp - 127 + 7;
    unsigned mag;
    if (e8 >= 16) {
        mag = 0x7E;
    } else if (e8 >= 1) {
        unsigned m = man >> 20;
        unsigned rem = man & 0xFFFFFu;
        if (rem > 0x80000u || (rem == 0x80000u && (m & 1))) m++;
        if (m == 8) { m = 0; e8++; }
        mag = (e8 >= 16) ? 0x7Eu : (((unsigned)e8 << 3) | m);
    } else {
        int shift = 20 + (1 - e8);
        if (shift > 24) {
            mag = 0;
        } else {
            unsigned full = 0x800000u | man;
            unsigned m = full >> shift;
            unsigned rem = full & ((1u << shift) - 1u);
            unsigned half = 1u << (shift - 1);
            if (rem > half || (rem == half && (m & 1))) m++;
            mag = (m > 7) ? 0x08u : m;
        }
    }
    return (unsigned char)(s | mag);
}

union FragU  { v16bf v; uint4 q[2]; };
union Frag8U { v8i_t v; uint2 q[4]; };

DEVFN v16bf load_a_frag(const unsigned short* A, int lda, int row0, int k0) {
    int lane = threadIdx.x & 31;
    int row  = row0 + (lane & 15);
    int kb   = k0 + ((lane & 16) ? 8 : 0);
    FragU f;
    f.q[0] = *(const uint4*)(A + (size_t)row * lda + kb);
    f.q[1] = *(const uint4*)(A + (size_t)row * lda + kb + 16);
    return f.v;
}
DEVFN v16bf load_b_frag(const unsigned short* Bp) {
    int lane = threadIdx.x & 31;
    FragU f;
    const uint4* p = (const uint4*)(Bp + (size_t)lane * 16);
    f.q[0] = p[0];
    f.q[1] = p[1];
    return f.v;
}
DEVFN v8i_t load_a8(const unsigned char* A, int lda, int row0, int k0) {
    int lane = threadIdx.x & 31;
    int row  = row0 + (lane & 15);
    const unsigned char* p = A + (size_t)row * lda + k0 + ((lane & 16) ? 8 : 0);
    Frag8U f;
    f.q[0] = *(const uint2*)(p);
    f.q[1] = *(const uint2*)(p + 16);
    f.q[2] = *(const uint2*)(p + 32);
    f.q[3] = *(const uint2*)(p + 48);
    return f.v;
}
DEVFN v8i_t load_b8(const unsigned char* Bp) {
    int lane = threadIdx.x & 31;
    const uint4* p = (const uint4*)(Bp + (size_t)lane * 32);
    uint4 a = p[0], b = p[1];
    Frag8U f;
    f.q[0].x = a.x; f.q[0].y = a.y;
    f.q[1].x = a.z; f.q[1].y = a.w;
    f.q[2].x = b.x; f.q[2].y = b.y;
    f.q[3].x = b.z; f.q[3].y = b.w;
    return f.v;
}
DEVFN v8f wmma_bf16(v16bf a, v16bf b, v8f c) {
    return __builtin_amdgcn_wmma_f32_16x16x32_bf16(false, a, false, b, (short)0, c, false, false);
}
DEVFN v8f wmma_fp8(v8i_t a, v8i_t b, v8f c) {
    return __builtin_amdgcn_wmma_f32_16x16x64_fp8_fp8(a, b, (short)0, c, false, false);
}
DEVFN v8f fill8(float x) {
    v8f r;
#pragma unroll
    for (int e = 0; e < 8; e++) r[e] = x;
    return r;
}


__global__ void pack_b_kernel(const float* __restrict__ W, int Ksrc, int Nrows,
                              unsigned short* __restrict__ dst, int Ntiles) {
    int tile = blockIdx.x;
    int kt = tile / Ntiles, nt = tile % Ntiles;
    int lane = threadIdx.x;
    int n  = nt * 16 + (lane & 15);
    int kb = kt * 32 + ((lane & 16) ? 8 : 0);
    unsigned short* o = dst + ((size_t)tile * 32 + lane) * 16;
#pragma unroll
    for (int h = 0; h < 16; h++) {
        int k = kb + (h < 8 ? h : 16 + (h - 8));
        float v = (k < Ksrc && n < Nrows) ? W[(size_t)n * Ksrc + k] : 0.0f;
        o[h] = f2bf(v);
    }
}

__global__ void pack_b8_kernel(const float* __restrict__ W, int Ksrc, int Nrows,
                               float scale, unsigned char* __restrict__ dst, int Ntiles) {
    int tile = blockIdx.x;
    int kt = tile / Ntiles, nt = tile % Ntiles;
    int lane = threadIdx.x;
    int n = nt * 16 + (lane & 15);
    unsigned char* o = dst + ((size_t)tile * 32 + lane) * 32;
#pragma unroll
    for (int c = 0; c < 4; c++) {
        int kstart = kt * 64 + c * 16 + ((lane & 16) ? 8 : 0);
#pragma unroll
        for (int b = 0; b < 8; b++) {
            int k = kstart + b;
            float v = (k < Ksrc && n < Nrows) ? W[(size_t)n * Ksrc + k] * scale : 0.0f;
            o[c * 8 + b] = f2fp8(v);
        }
    }
}

__global__ void add_bias_kernel(const float* a, const float* b, float* o, int n) {
    int i = blockIdx.x * 256 + threadIdx.x;
    if (i < n) o[i] = a[i] + b[i];
}

__global__ void f32_to_bf_kernel(const float* src, unsigned short* dst, int n) {
    int i = blockIdx.x * 256 + threadIdx.x;
    if (i < n) dst[i] = f2bf(src[i]);
}

__global__ void build_ns_kernel(const float* __restrict__ state,
                                const float* __restrict__ action,
                                unsigned short* __restrict__ ns) {
    int idx = blockIdx.x * 256 + threadIdx.x;
    int k = idx & 31, t = (idx >> 5) & 63, b = idx >> 11;
    float v = 0.0f;
    if (k < 7)       v = state[(size_t)b * 448 + t * 7 + k];
    else if (k < 16) v = action[(size_t)b * 9 + (k - 7)];
    ns[idx] = f2bf(v);
}

__global__ void build_ns8_kernel(const float* __restrict__ state,
                                 const float* __restrict__ action,
                                 unsigned char* __restrict__ ns8) {
    int idx = blockIdx.x * 256 + threadIdx.x;
    int k = idx & 15, t = (idx >> 4) & 63, b = idx >> 10;
    float v = (k < 7) ? state[(size_t)b * 448 + t * 7 + k]
                      : action[(size_t)b * 9 + (k - 7)];
    ns8[idx] = f2fp8(v);
}

__global__ void copy_h_to_abuf_kernel(const unsigned short* hbf, unsigned short* abuf) {
    size_t i = (size_t)blockIdx.x * 256 + threadIdx.x;
    int j = (int)(i & 255);
    size_t rb = i >> 8;
    abuf[rb * 768 + 512 + j] = hbf[rb * 256 + j];
}

template<int KT, int NT, int WAVES, bool OUTBF, bool RELU>
__global__ void gemm_bf16_kernel(const unsigned short* __restrict__ A, int lda,
                                 const unsigned short* __restrict__ Bp,
                                 const float* __restrict__ biascol,
                                 const float* __restrict__ rowscale,
                                 void* __restrict__ Cv, int ldc) {
    constexpr int NPW = NT / WAVES;
    int wave = threadIdx.x >> 5, lane = threadIdx.x & 31;
    int r0 = blockIdx.x * 32;
    v8f acc[2][NPW];
#pragma unroll
    for (int mt = 0; mt < 2; mt++)
#pragma unroll
        for (int i = 0; i < NPW; i++) acc[mt][i] = fill8(0.0f);

    for (int kt = 0; kt < KT; kt++) {
        if (kt + 1 < KT)
            __builtin_prefetch(Bp + ((size_t)((kt + 1) * NT + wave * NPW) << 9), 0, 3);
        v16bf a0 = load_a_frag(A, lda, r0,      kt * 32);
        v16bf a1 = load_a_frag(A, lda, r0 + 16, kt * 32);
#pragma unroll
        for (int i = 0; i < NPW; i++) {
            int nt = wave * NPW + i;
            v16bf bb = load_b_frag(Bp + ((size_t)(kt * NT + nt) << 9));
            acc[0][i] = wmma_bf16(a0, bb, acc[0][i]);
            acc[1][i] = wmma_bf16(a1, bb, acc[1][i]);
        }
    }
#pragma unroll
    for (int mt = 0; mt < 2; mt++)
#pragma unroll
        for (int i = 0; i < NPW; i++) {
            int nt  = wave * NPW + i;
            int col = nt * 16 + (lane & 15);
            float bias = biascol ? biascol[col] : 0.0f;
#pragma unroll
            for (int e = 0; e < 8; e++) {
                int r = r0 + mt * 16 + ((lane & 16) ? 8 + e : e);
                float v = acc[mt][i][e] + bias * (rowscale ? rowscale[r] : 1.0f);
                if (RELU) v = fmaxf(v, 0.0f);
                if (OUTBF) ((unsigned short*)Cv)[(size_t)r * ldc + col] = f2bf(v);
                else       ((float*)Cv)[(size_t)r * ldc + col] = v;
            }
        }
}

__global__ __launch_bounds__(512, 1)
void serial_enc_fp8_kernel(const unsigned char* __restrict__ ns8,
                           const unsigned char* __restrict__ packS8,
                           const float* __restrict__ bsum,
                           const float* __restrict__ h1, const float* __restrict__ c1,
                           const float* __restrict__ h2, const float* __restrict__ c2,
                           unsigned short* __restrict__ abuf) {
    int enc = blockIdx.y;
    const float* h0 = enc ? h2 : h1;
    const float* c0 = enc ? c2 : c1;
    const unsigned char* Bp = packS8 + (size_t)enc * 5 * 64 * 1024;
    const float* bias = bsum + enc * 1024;
    int b0 = blockIdx.x * 16;
    int wave = threadIdx.x >> 5, lane = threadIdx.x & 31;
    int j = wave * 16 + (lane & 15);

    __shared__ unsigned char Ast[16][336];

    for (int i = threadIdx.x; i < 16 * 256; i += 512) {
        int r = i >> 8, jj = i & 255;
        Ast[r][64 + jj] = f2fp8(h0[(size_t)(b0 + r) * 256 + jj]);
    }
    for (int i = threadIdx.x; i < 16 * 48; i += 512) {
        int r = i / 48, k = i % 48;
        Ast[r][16 + k] = 0;
    }
    float creg[8];
#pragma unroll
    for (int e = 0; e < 8; e++) {
        int r = b0 + ((lane & 16) ? 8 + e : e);
        creg[e] = c0[(size_t)r * 256 + j];
    }
    float bq[4];
#pragma unroll
    for (int q = 0; q < 4; q++) bq[q] = bias[q * 256 + j] * 16.0f;

    v8i_t Breg[5][4];
#pragma unroll
    for (int kt = 0; kt < 5; kt++)
#pragma unroll
        for (int q = 0; q < 4; q++)
            Breg[kt][q] = load_b8(Bp + ((size_t)(kt * 64 + q * 16 + wave) << 10));
    __syncthreads();

    for (int t = 63; t >= 1; --t) {
        if (threadIdx.x < 256) {
            int r = threadIdx.x >> 4, k = threadIdx.x & 15;
            Ast[r][k] = ns8[((size_t)(b0 + r) * 64 + t) * 16 + k];
        }
        __syncthreads();

        v8f acc[4];
#pragma unroll
        for (int q = 0; q < 4; q++) acc[q] = fill8(bq[q]);

#pragma unroll
        for (int kt = 0; kt < 5; kt++) {
            v8i_t a = load_a8(&Ast[0][0], 336, 0, kt * 64);
#pragma unroll
            for (int q = 0; q < 4; q++)
                acc[q] = wmma_fp8(a, Breg[kt][q], acc[q]);
        }
        __syncthreads();

#pragma unroll
        for (int e = 0; e < 8; e++) {
            float i_ = sigm(acc[0][e] * 0.0625f);
            float f_ = sigm(acc[1][e] * 0.0625f);
            float g_ = tanhf(acc[2][e] * 0.0625f);
            float o_ = sigm(acc[3][e] * 0.0625f);
            float cc = f_ * creg[e] + i_ * g_;
            creg[e] = cc;
            float hh = o_ * tanhf(cc);
            int r = (lane & 16) ? 8 + e : e;
            Ast[r][64 + j] = f2fp8(hh);
            if (t == 1)
                abuf[((size_t)enc * 1024 + b0 + r) * 768 + j] = f2bf(hh);
        }
        __syncthreads();
    }
}

__global__ __launch_bounds__(512, 1)
void parallel_gates_kernel(const float* __restrict__ state,
                           const unsigned short* __restrict__ ns,
                           const unsigned short* __restrict__ packPWih,
                           const float* __restrict__ hterm,
                           const float* __restrict__ c1, const float* __restrict__ c2,
                           unsigned short* __restrict__ ubuf,
                           float* __restrict__ S0buf, float* __restrict__ S1buf) {
    int enc = blockIdx.y, b = blockIdx.x;
    const float* c0 = enc ? c2 : c1;
    const unsigned short* Bp = packPWih + (size_t)enc * 64 * 512;
    const float* ht = hterm + ((size_t)enc * 1024 + b) * 1024;
    int wave = threadIdx.x >> 5, lane = threadIdx.x & 31;
    int j = wave * 16 + (lane & 15);

    __shared__ float s0s[64], s1s[64];
    __shared__ float Ss[2];
    if (threadIdx.x < 64) {
        s0s[threadIdx.x] = state[(size_t)b * 448 + threadIdx.x * 7] * 0.1f;
        s1s[threadIdx.x] = state[(size_t)b * 448 + threadIdx.x * 7 + 1];
    }
    __syncthreads();
    if (threadIdx.x < 2) {
        const float* src = threadIdx.x ? s1s : s0s;
        float s = 0.0f;
        for (int t = 0; t < 64; t++) s += src[t];
        Ss[threadIdx.x] = s;
    }

    v8f acc[4][4];
#pragma unroll
    for (int q = 0; q < 4; q++) {
        float hv = ht[q * 256 + j];
        v8f x = fill8(hv);
#pragma unroll
        for (int mt = 0; mt < 4; mt++) acc[mt][q] = x;
    }

    const unsigned short* Ab = ns + ((size_t)b * 64 + 1) * 32;
    v16bf afr[4];
#pragma unroll
    for (int mt = 0; mt < 4; mt++) afr[mt] = load_a_frag(Ab, 32, mt * 16, 0);
#pragma unroll
    for (int q = 0; q < 4; q++) {
        v16bf bb = load_b_frag(Bp + ((size_t)(q * 16 + wave) << 9));
#pragma unroll
        for (int mt = 0; mt < 4; mt++)
            acc[mt][q] = wmma_bf16(afr[mt], bb, acc[mt][q]);
    }
    __syncthreads();

    float c0j = c0[(size_t)b * 256 + j];
    float u1 = 0.0f, u2 = 0.0f;
#pragma unroll
    for (int mt = 0; mt < 4; mt++)
#pragma unroll
        for (int e = 0; e < 8; e++) {
            int m = mt * 16 + ((lane & 16) ? 8 + e : e);
            int t = m + 1;
            if (t <= 63) {
                float i_ = sigm(acc[mt][0][e]);
                float f_ = sigm(acc[mt][1][e]);
                float g_ = tanhf(acc[mt][2][e]);
                float o_ = sigm(acc[mt][3][e]);
                float cc = f_ * c0j + i_ * g_;
                float hh = o_ * tanhf(cc);
                u1 += s0s[t] * hh;
                u2 += s1s[t] * hh;
            }
        }
    u1 += __shfl_xor(u1, 16);
    u2 += __shfl_xor(u2, 16);
    if (lane < 16) {
        ubuf[((size_t)enc * 1024 + b) * 512 + j]       = f2bf(u1);
        ubuf[((size_t)enc * 1024 + b) * 512 + 256 + j] = f2bf(u2);
    }
    if (enc == 0 && threadIdx.x == 0) { S0buf[b] = Ss[0]; S1buf[b] = Ss[1]; }
}

__global__ void lstm_head_kernel(const unsigned short* __restrict__ abuf,
                                 const unsigned short* __restrict__ packL,
                                 const float* __restrict__ biasL,
                                 const float* __restrict__ c1, const float* __restrict__ c2,
                                 unsigned short* __restrict__ qbuf) {
    int enc = blockIdx.y;
    const float* c0 = enc ? c2 : c1;
    const unsigned short* A = abuf + (size_t)enc * 1024 * 768;
    int b0 = blockIdx.x * 32;
    int wave = threadIdx.x >> 5, lane = threadIdx.x & 31;

    v8f acc[2][8];
#pragma unroll
    for (int q = 0; q < 4; q++)
#pragma unroll
        for (int p = 0; p < 2; p++) {
            float bv = biasL[q * 256 + (2 * wave + p) * 16 + (lane & 15)];
            acc[0][q * 2 + p] = fill8(bv);
            acc[1][q * 2 + p] = fill8(bv);
        }
    for (int kt = 0; kt < 24; kt++) {
        if (kt + 1 < 24)
            __builtin_prefetch(packL + ((size_t)((kt + 1) * 64 + 2 * wave) << 9), 0, 3);
        v16bf a0 = load_a_frag(A, 768, b0,      kt * 32);
        v16bf a1 = load_a_frag(A, 768, b0 + 16, kt * 32);
#pragma unroll
        for (int q = 0; q < 4; q++)
#pragma unroll
            for (int p = 0; p < 2; p++) {
                int nt = q * 16 + 2 * wave + p;
                v16bf bb = load_b_frag(packL + ((size_t)(kt * 64 + nt) << 9));
                acc[0][q * 2 + p] = wmma_bf16(a0, bb, acc[0][q * 2 + p]);
                acc[1][q * 2 + p] = wmma_bf16(a1, bb, acc[1][q * 2 + p]);
            }
    }
#pragma unroll
    for (int mt = 0; mt < 2; mt++)
#pragma unroll
        for (int p = 0; p < 2; p++)
#pragma unroll
            for (int e = 0; e < 8; e++) {
                int r = b0 + mt * 16 + ((lane & 16) ? 8 + e : e);
                int j = (2 * wave + p) * 16 + (lane & 15);
                float i_ = sigm(acc[mt][0 + p][e]);
                float f_ = sigm(acc[mt][2 + p][e]);
                float g_ = tanhf(acc[mt][4 + p][e]);
                float o_ = sigm(acc[mt][6 + p][e]);
                float cc = f_ * c0[(size_t)r * 256 + j] + i_ * g_;
                float hh = o_ * tanhf(cc);
                qbuf[((size_t)enc * 1024 + r) * 256 + j] = f2bf(hh);
            }
}

__global__ void dot_head_kernel(const unsigned short* __restrict__ R,
                                const float* __restrict__ w,
                                const float* __restrict__ bias,
                                float* __restrict__ out) {
    int wv = threadIdx.x >> 5, lane = threadIdx.x & 31;
    int b = blockIdx.x * 8 + wv;
    const unsigned short* r = R + (size_t)b * 256;
    float s = 0.0f;
    for (int k = lane; k < 256; k += 32) s += bf2f(r[k]) * w[k];
#pragma unroll
    for (int off = 16; off; off >>= 1) s += __shfl_xor(s, off);
    if (lane == 0) out[b] = s + bias[0];
}

extern "C" void kernel_launch(void* const* d_in, const int* in_sizes, int n_in,
                              void* d_out, int out_size, void* d_ws, size_t ws_size,
                              hipStream_t stream) {
    (void)in_sizes; (void)n_in; (void)out_size; (void)ws_size;
    auto F = [&](int i) { return (const float*)d_in[i]; };

    char* base = (char*)d_ws; size_t off = 0;
    auto alloc = [&](size_t bytes) -> void* {
        void* p = base + off;
        off = (off + bytes + 255) & ~(size_t)255;
        return p;
    };
    unsigned short* ns       = (unsigned short*)alloc((size_t)(1024 * 64 * 32 + 64) * 2);
    unsigned char*  ns8      = (unsigned char*) alloc((size_t)1024 * 64 * 16);
    unsigned char*  packS8   = (unsigned char*) alloc((size_t)2 * 5 * 64 * 1024);
    unsigned short* packPWih = (unsigned short*)alloc((size_t)2 * 64 * 512 * 2);
    unsigned short* packPWhh = (unsigned short*)alloc((size_t)2 * 8 * 64 * 512 * 2);
    unsigned short* packL    = (unsigned short*)alloc((size_t)24 * 64 * 512 * 2);
    unsigned short* packF    = (unsigned short*)alloc((size_t)4 * 8 * 8 * 512 * 2);
    unsigned short* packl2   = (unsigned short*)alloc((size_t)2 * 8 * 16 * 512 * 2);
    float*          bsum     = (float*)alloc((size_t)5 * 1024 * 4);
    unsigned short* hbf      = (unsigned short*)alloc((size_t)2 * 1024 * 256 * 2);
    float*          hterm    = (float*)alloc((size_t)2 * 1024 * 1024 * 4);
    unsigned short* abuf     = (unsigned short*)alloc((size_t)2 * 1024 * 768 * 2);
    unsigned short* ubuf     = (unsigned short*)alloc((size_t)2 * 1024 * 512 * 2);
    float*          S0buf    = (float*)alloc(1024 * 4);
    float*          S1buf    = (float*)alloc(1024 * 4);
    unsigned short* qbuf     = (unsigned short*)alloc((size_t)2 * 1024 * 256 * 2);
    unsigned short* rbuf     = (unsigned short*)alloc((size_t)2 * 1024 * 256 * 2);

    build_ns_kernel<<<8192, 256, 0, stream>>>(F(0), F(1), ns);
    build_ns8_kernel<<<4096, 256, 0, stream>>>(F(0), F(1), ns8);
    f32_to_bf_kernel<<<1024, 256, 0, stream>>>(F(2), hbf, 262144);
    f32_to_bf_kernel<<<1024, 256, 0, stream>>>(F(4), hbf + 262144, 262144);
    copy_h_to_abuf_kernel<<<2048, 256, 0, stream>>>(hbf, abuf);

    add_bias_kernel<<<4, 256, 0, stream>>>(F(13), F(12), bsum,        1024);
    add_bias_kernel<<<4, 256, 0, stream>>>(F(17), F(16), bsum + 1024, 1024);
    add_bias_kernel<<<4, 256, 0, stream>>>(F(23), F(22), bsum + 2048, 1024);
    add_bias_kernel<<<4, 256, 0, stream>>>(F(33), F(32), bsum + 3072, 1024);
    add_bias_kernel<<<4, 256, 0, stream>>>(F(9),  F(8),  bsum + 4096, 1024);

    pack_b8_kernel<<<64,  32, 0, stream>>>(F(11), 16,  1024, 16.0f, packS8,                  64);
    pack_b8_kernel<<<256, 32, 0, stream>>>(F(10), 256, 1024, 16.0f, packS8 + 65536,          64);
    pack_b8_kernel<<<64,  32, 0, stream>>>(F(15), 16,  1024, 16.0f, packS8 + 327680,         64);
    pack_b8_kernel<<<256, 32, 0, stream>>>(F(14), 256, 1024, 16.0f, packS8 + 327680 + 65536, 64);
    pack_b_kernel<<<64,  32, 0, stream>>>(F(21), 16,  1024, packPWih,          64);
    pack_b_kernel<<<64,  32, 0, stream>>>(F(31), 16,  1024, packPWih + 32768,  64);
    pack_b_kernel<<<512, 32, 0, stream>>>(F(20), 256, 1024, packPWhh,          64);
    pack_b_kernel<<<512, 32, 0, stream>>>(F(30), 256, 1024, packPWhh + 262144, 64);
    pack_b_kernel<<<1024, 32, 0, stream>>>(F(7), 512, 1024, packL,          64);
    pack_b_kernel<<<512,  32, 0, stream>>>(F(6), 256, 1024, packL + 524288, 64);
    pack_b_kernel<<<64, 32, 0, stream>>>(F(24), 256, 128, packF,          8);
    pack_b_kernel<<<64, 32, 0, stream>>>(F(26), 256, 128, packF + 32768,  8);
    pack_b_kernel<<<64, 32, 0, stream>>>(F(34), 256, 128, packF + 65536,  8);
    pack_b_kernel<<<64, 32, 0, stream>>>(F(36), 256, 128, packF + 98304,  8);
    pack_b_kernel<<<128, 32, 0, stream>>>(F(18), 256, 256, packl2,         16);
    pack_b_kernel<<<128, 32, 0, stream>>>(F(38), 256, 256, packl2 + 65536, 16);

    serial_enc_fp8_kernel<<<dim3(64, 2), 512, 0, stream>>>(ns8, packS8, bsum,
                                                           F(2), F(3), F(4), F(5), abuf);

    gemm_bf16_kernel<8, 64, 8, false, false><<<32, 256, 0, stream>>>(
        hbf, 256, packPWhh, bsum + 2048, nullptr, hterm, 1024);
    gemm_bf16_kernel<8, 64, 8, false, false><<<32, 256, 0, stream>>>(
        hbf + 262144, 256, packPWhh + 262144, bsum + 3072, nullptr,
        hterm + (size_t)1024 * 1024, 1024);

    parallel_gates_kernel<<<dim3(1024, 2), 512, 0, stream>>>(
        F(0), ns, packPWih, hterm, F(3), F(5), ubuf, S0buf, S1buf);

    gemm_bf16_kernel<8, 8, 4, true, false><<<32, 128, 0, stream>>>(
        ubuf, 512, packF, F(25), S0buf, (void*)(abuf + 256), 768);
    gemm_bf16_kernel<8, 8, 4, true, false><<<32, 128, 0, stream>>>(
        ubuf + 256, 512, packF + 32768, F(27), S1buf, (void*)(abuf + 384), 768);
    gemm_bf16_kernel<8, 8, 4, true, false><<<32, 128, 0, stream>>>(
        ubuf + 524288, 512, packF + 65536, F(35), S0buf,
        (void*)(abuf + 786432 + 256), 768);
    gemm_bf16_kernel<8, 8, 4, true, false><<<32, 128, 0, stream>>>(
        ubuf + 524288 + 256, 512, packF + 98304, F(37), S1buf,
        (void*)(abuf + 786432 + 384), 768);

    lstm_head_kernel<<<dim3(32, 2), 256, 0, stream>>>(abuf, packL, bsum + 4096,
                                                      F(3), F(5), qbuf);
    gemm_bf16_kernel<8, 16, 8, true, true><<<32, 256, 0, stream>>>(
        qbuf, 256, packl2, F(19), nullptr, (void*)rbuf, 256);
    gemm_bf16_kernel<8, 16, 8, true, true><<<32, 256, 0, stream>>>(
        qbuf + 262144, 256, packl2 + 65536, F(39), nullptr, (void*)(rbuf + 262144), 256);

    dot_head_kernel<<<128, 256, 0, stream>>>(rbuf, F(28), F(29), (float*)d_out);
    dot_head_kernel<<<128, 256, 0, stream>>>(rbuf + 262144, F(40), F(41),
                                             (float*)d_out + 1024);
}
